// CTCDecoder_76106820485575
// MI455X (gfx1250) — matrix operand order repaired and checked by reading, not compiled, not run on hardware
//
#include <hip/hip_runtime.h>
#include <hip/hip_bf16.h>
#include <math.h>
#include <stdint.h>

// Problem constants (from reference)
static constexpr int CB = 32;     // batch
static constexpr int CT = 1024;   // time
static constexpr int CH = 512;    // hidden
static constexpr int CV = 1024;   // vocab
static constexpr int CS = 128;    // target len
static constexpr int CL = 2 * CS + 1;  // 257 extended states
#define NEGV (-1e30f)

typedef __bf16 bf16;
typedef __attribute__((ext_vector_type(16))) __bf16 v16bf;
typedef __attribute__((ext_vector_type(8)))  __bf16 v8bf;
typedef __attribute__((ext_vector_type(8)))  float  v8f;

// ---------------- conversion kernels ----------------
__global__ void cvt_f32_bf16(const float* __restrict__ in, bf16* __restrict__ o, int n) {
  int i = blockIdx.x * blockDim.x + threadIdx.x;
  if (i < n) o[i] = (bf16)in[i];
}

// W: [K][N] row-major  ->  Wt: [N][K] bf16 (N-major so B-fragments are contiguous in K)
__global__ void transpose_to_bf16(const float* __restrict__ W, bf16* __restrict__ Wt,
                                  int K, int N) {
  int i = blockIdx.x * blockDim.x + threadIdx.x;
  if (i < K * N) {
    int n = i / K, k = i % K;
    Wt[i] = (bf16)W[(size_t)k * N + n];
  }
}

// -------- async global->LDS copy helpers (CDNA5 ASYNCcnt path) --------
__device__ __forceinline__ void async_copy_b128(uint32_t lds_addr, const void* gaddr) {
  asm volatile("global_load_async_to_lds_b128 %0, %1, off"
               :
               : "v"(lds_addr), "v"((uint64_t)(uintptr_t)gaddr)
               : "memory");
}
__device__ __forceinline__ void wait_async_le4() {
  asm volatile("s_wait_asynccnt 0x4" ::: "memory");
}
__device__ __forceinline__ void wait_async_0() {
  asm volatile("s_wait_asynccnt 0x0" ::: "memory");
}

// ---------------- bf16 WMMA GEMM, 128x128x32 tiles, 8 waves ----------------
// Double-buffered LDS with async global->LDS DMA; one barrier per K-step.
// MODE 0: out = tanh(acc + bias) -> bf16 ; MODE 1: out = acc + bias -> f32
template <int MODE>
__global__ __launch_bounds__(256) void gemm_wmma(
    const bf16* __restrict__ A,    // [M][K] row-major bf16
    const bf16* __restrict__ Bt,   // [N][K] bf16 (transposed weights)
    const float* __restrict__ bias,
    bf16* __restrict__ outb, float* __restrict__ outf,
    int N, int K) {
  __shared__ bf16 As[2][128 * 32];
  __shared__ bf16 Bs[2][128 * 32];
  constexpr uint32_t BUFB = 128 * 32 * sizeof(bf16);  // 8 KB per buffer

  const int tid   = threadIdx.x;
  const int lane  = tid & 31;
  const int wave  = tid >> 5;          // 0..7
  const int wm    = (wave >> 1) * 32;  // wave M offset (4 rows of waves)
  const int wn    = (wave & 1) * 64;   // wave N offset (2 cols of waves)
  const int lrow  = lane & 15;         // M (A) or N-column (B/C) within fragment
  const int lhalf = lane >> 4;         // K-half / M-half selector

  const int mBase = blockIdx.y * 128;
  const int nBase = blockIdx.x * 128;

  const int ldRow  = tid >> 1;  // 0..127 : tile row this thread stages
  const int ldPart = tid & 1;   // 0/1    : which 16-element half of the 32-wide row

  // LDS byte addresses (low 32 bits of the flat shared address = LDS offset)
  const uint32_t ldsA0 = (uint32_t)(uintptr_t)&As[0][ldRow * 32 + ldPart * 16];
  const uint32_t ldsB0 = (uint32_t)(uintptr_t)&Bs[0][ldRow * 32 + ldPart * 16];
  const bf16* gA = A  + (size_t)(mBase + ldRow) * K + ldPart * 16;
  const bf16* gB = Bt + (size_t)(nBase + ldRow) * K + ldPart * 16;

  auto issue_tile = [&](int buf, int k0) {
    const uint32_t la = ldsA0 + (uint32_t)buf * BUFB;
    const uint32_t lb = ldsB0 + (uint32_t)buf * BUFB;
    async_copy_b128(la,      gA + k0);      // 16 B
    async_copy_b128(la + 16, gA + k0 + 8);  // 16 B
    async_copy_b128(lb,      gB + k0);
    async_copy_b128(lb + 16, gB + k0 + 8);
  };

  v8f acc[2][4] = {};

  issue_tile(0, 0);  // prologue prefetch

  int buf = 0;
  for (int k0 = 0; k0 < K; k0 += 32, buf ^= 1) {
    const bool hasNext = (k0 + 32 < K);
    if (hasNext) issue_tile(buf ^ 1, k0 + 32);   // prefetch next tile (4 ops in flight)
    if (hasNext) wait_async_le4();               // in-order: current buffer's 4 ops done
    else         wait_async_0();
    __syncthreads();                             // tile visible to all waves

    // A fragment (16x32): lane holds row lrow,
    // K = lhalf*16+{0..15} (matches B)  -> two 16B ds loads.
    v16bf af[2];
#pragma unroll
    for (int fm = 0; fm < 2; ++fm) {
      union { v16bf v; struct { v8bf lo, hi; } p; } u;
      const bf16* base = &As[buf][(wm + fm * 16 + lrow) * 32 + lhalf * 16];
      u.p.lo = *(const v8bf*)base;
      u.p.hi = *(const v8bf*)(base + 8);
      af[fm] = u.v;
    }
    // B fragment (32x16): lane holds column lrow, K = lhalf*16+{0..15},
    // contiguous in the N-major LDS tile -> one 32B ds load.
    v16bf bfv[4];
#pragma unroll
    for (int fn = 0; fn < 4; ++fn)
      bfv[fn] = *(const v16bf*)&Bs[buf][(wn + fn * 16 + lrow) * 32 + lhalf * 16];

#pragma unroll
    for (int fm = 0; fm < 2; ++fm)
#pragma unroll
      for (int fn = 0; fn < 4; ++fn)
        acc[fm][fn] = __builtin_amdgcn_wmma_f32_16x16x32_bf16(
            false, af[fm], false, bfv[fn], (short)0, acc[fm][fn], false, false);
    // No trailing barrier: each wave's ds_loads complete before its WMMAs,
    // which precede the next iteration's barrier; buffer `buf` is rewritten
    // by the next iteration's prefetch, issued before that barrier.
  }

  // Epilogue. C/D layout: element r -> row (lane/16)*8 + r, col lane%16.
#pragma unroll
  for (int fm = 0; fm < 2; ++fm)
#pragma unroll
    for (int fn = 0; fn < 4; ++fn) {
      const int gr0 = mBase + wm + fm * 16 + lhalf * 8;
      const int gc  = nBase + wn + fn * 16 + lrow;
      const float bv = bias[gc];
      v8f c = acc[fm][fn];
#pragma unroll
      for (int r = 0; r < 8; ++r) {
        float v = c[r] + bv;
        if (MODE == 0) outb[(size_t)(gr0 + r) * N + gc] = (bf16)tanhf(v);
        else           outf[(size_t)(gr0 + r) * N + gc] = v;
      }
    }
}

// ---------------- rowwise logsumexp over V (one wave per row) ----------------
__global__ __launch_bounds__(256) void row_lse(const float* __restrict__ logits,
                                               float* __restrict__ lse) {
  const int lane = threadIdx.x & 31;
  const int row  = blockIdx.x * 8 + (threadIdx.x >> 5);
  const float4* p = (const float4*)(logits + (size_t)row * CV);
  float mx = -INFINITY;
  for (int v = lane; v < CV / 4; v += 32) {
    float4 q = p[v];
    mx = fmaxf(mx, fmaxf(fmaxf(q.x, q.y), fmaxf(q.z, q.w)));
  }
  for (int off = 16; off; off >>= 1) mx = fmaxf(mx, __shfl_xor(mx, off, 32));
  float s = 0.f;
  for (int v = lane; v < CV / 4; v += 32) {
    float4 q = p[v];
    s += expf(q.x - mx) + expf(q.y - mx) + expf(q.z - mx) + expf(q.w - mx);
  }
  for (int off = 16; off; off >>= 1) s += __shfl_xor(s, off, 32);
  if (lane == 0) lse[row] = mx + logf(s);
}

// ---------------- CTC alpha recursion ----------------
__device__ __forceinline__ float lae(float a, float b) {
  float mx = fmaxf(a, b), mn = fminf(a, b);
  if (mx <= -1e29f) return NEGV;
  return mx + log1pf(expf(mn - mx));
}

__global__ __launch_bounds__(288) void ctc_alpha(
    const float* __restrict__ logits, const float* __restrict__ lse,
    const int* __restrict__ targets, const int* __restrict__ in_len,
    float* __restrict__ out) {
  __shared__ float bufA[CL], bufB[CL];
  __shared__ int ext_sh[CL];
  __shared__ int tl_sh;

  const int b = blockIdx.x;
  const int s = threadIdx.x;
  const bool valid = s < CL;

  if (threadIdx.x == 0) {
    int tl = 0;
    for (int i = 0; i < CS; ++i) tl += (targets[b * CS + i] != 0);
    tl_sh = tl;
  }
  int ext = 0;
  if (valid) {
    ext = (s & 1) ? targets[b * CS + (s >> 1)] : 0;  // blank = 0
    ext_sh[s] = ext;
  }
  __syncthreads();
  const bool skip = valid && s >= 2 && ext != 0 && ext != ext_sh[s - 2];
  const int Tin = in_len[b];
  const size_t rowbase = (size_t)b * CT;

  float* cur = bufA;
  float* nxt = bufB;
  if (valid)
    cur[s] = (s < 2) ? (logits[rowbase * CV + ext] - lse[rowbase]) : NEGV;
  __syncthreads();

  for (int t = 1; t < CT; ++t) {
    if (valid) {
      float a = cur[s];
      if (t < Tin) {
        float v = a;
        if (s >= 1) v = lae(v, cur[s - 1]);
        if (skip)   v = lae(v, cur[s - 2]);
        a = v + (logits[(rowbase + t) * CV + ext] - lse[rowbase + t]);
      }
      nxt[s] = a;
    }
    __syncthreads();
    float* tmp = cur; cur = nxt; nxt = tmp;
  }

  if (threadIdx.x == 0) {
    int tl = tl_sh;
    float l1 = cur[2 * tl - 1];
    float l2 = cur[2 * tl];
    out[b] = -lae(l1, l2) / (float)tl;
  }
}

// ---------------- launch ----------------
extern "C" void kernel_launch(void* const* d_in, const int* in_sizes, int n_in,
                              void* d_out, int out_size, void* d_ws, size_t ws_size,
                              hipStream_t stream) {
  const int*   tgt  = (const int*)d_in[0];
  const float* X    = (const float*)d_in[1];
  const int*   ilen = (const int*)d_in[2];
  const float* W1   = (const float*)d_in[3];
  const float* b1   = (const float*)d_in[4];
  const float* W2   = (const float*)d_in[5];
  const float* b2   = (const float*)d_in[6];
  float* out = (float*)d_out;

  char* ws = (char*)d_ws;
  size_t off = 0;
  auto alloc = [&](size_t bytes) -> void* {
    void* p = ws + off;
    off = (off + bytes + 255) & ~(size_t)255;
    return p;
  };
  bf16*  Xb     = (bf16*)alloc((size_t)CB * CT * CH * sizeof(bf16));
  bf16*  W1t    = (bf16*)alloc((size_t)CH * CH * sizeof(bf16));
  bf16*  W2t    = (bf16*)alloc((size_t)CV * CH * sizeof(bf16));
  bf16*  Hb     = (bf16*)alloc((size_t)CB * CT * CH * sizeof(bf16));
  float* logits = (float*)alloc((size_t)CB * CT * CV * sizeof(float));
  float* lseArr = (float*)alloc((size_t)CB * CT * sizeof(float));

  const int M = CB * CT;  // 32768 rows

  int nX = M * CH;
  cvt_f32_bf16<<<(nX + 255) / 256, 256, 0, stream>>>(X, Xb, nX);
  transpose_to_bf16<<<(CH * CH + 255) / 256, 256, 0, stream>>>(W1, W1t, CH, CH);
  transpose_to_bf16<<<(CH * CV + 255) / 256, 256, 0, stream>>>(W2, W2t, CH, CV);

  dim3 g1(CH / 128, M / 128);  // 4 x 256 blocks
  gemm_wmma<0><<<g1, 256, 0, stream>>>(Xb, W1t, b1, Hb, nullptr, CH, CH);

  dim3 g2(CV / 128, M / 128);  // 8 x 256 blocks
  gemm_wmma<1><<<g2, 256, 0, stream>>>(Hb, W2t, b2, nullptr, logits, CV, CH);

  row_lse<<<M / 8, 256, 0, stream>>>(logits, lseArr);
  ctc_alpha<<<CB, 288, 0, stream>>>(logits, lseArr, tgt, ilen, out);
}
